// CTCrackSeg_6408091205907
// MI455X (gfx1250) — matrix operand order repaired and checked by reading, not compiled, not run on hardware
//
#include <hip/hip_runtime.h>
#include <hip/hip_bf16.h>

// ---------------------------------------------------------------------------
// Deformable conv (B=2, C=64, H=W=256, KS=3) for MI455X / gfx1250.
//   k_pad_nhwc : x NCHW f32 -> padded (258x258) NHWC bf16  (gathers contiguous)
//   k_prep_w   : weights -> per-lane WMMA B-fragment order (bf16)
//   k_offconv  : implicit-GEMM 3x3 conv 64->18(+pad 32), async-LDS staged A
//   k_main     : fused bilinear sampling + per-pixel GEMM (M=32,K=576,N=64)
// bf16 WMMA with f32 accumulate everywhere; offsets kept f32 for bilinear
// precision. A-tiles staged in LDS; B-fragments streamed from L2-resident
// pre-swizzled buffers (108 KB total, reused by every workgroup).
// ---------------------------------------------------------------------------

typedef __attribute__((ext_vector_type(16))) __bf16 v16bf;
typedef __attribute__((ext_vector_type(8)))  float  v8f;
typedef __attribute__((ext_vector_type(4)))  int    v4i;

#define HH    256
#define WW    256
#define CIN   64
#define HP    258
#define WP    258
#define KTOT  576   // 64 ch * 9 taps
#define KT    18    // 576/32 K-steps

union Frag { uint4 q[2]; v16bf v; };

// ---- CDNA5 async global->LDS path (guarded; falls back to VGPR copies) ----
#if defined(__HIP_DEVICE_COMPILE__) && __has_builtin(__builtin_amdgcn_global_load_async_to_lds_b128)
#define HAVE_ASYNC 1
#else
#define HAVE_ASYNC 0
#endif

static __device__ __forceinline__ void async_cp16(const void* g, void* l) {
#if HAVE_ASYNC
  __builtin_amdgcn_global_load_async_to_lds_b128((v4i*)g, (v4i*)l, 0, 0);
#else
  *(uint4*)l = *(const uint4*)g;
#endif
}
static __device__ __forceinline__ void wait_async0() {
#if HAVE_ASYNC
#if __has_builtin(__builtin_amdgcn_s_wait_asynccnt)
  __builtin_amdgcn_s_wait_asynccnt(0);
#else
  asm volatile("s_wait_asynccnt 0" ::: "memory");
#endif
#endif
}

static __device__ __forceinline__ unsigned short f2bf(float f) {
  unsigned u = __builtin_bit_cast(unsigned, f);
  u += 0x7FFFu + ((u >> 16) & 1u);          // round-to-nearest-even
  return (unsigned short)(u >> 16);
}
static __device__ __forceinline__ float bf2f(unsigned h) {
  unsigned u = (h & 0xFFFFu) << 16;
  return __builtin_bit_cast(float, u);
}
static __device__ __forceinline__ unsigned pack2(float a, float b) {
  return (unsigned)f2bf(a) | ((unsigned)f2bf(b) << 16);
}

// --------------------------------------------------------------------------
// Kernel 1: NCHW f32 -> padded NHWC bf16 (zero border). 2*258*258*64 elems.
// --------------------------------------------------------------------------
__global__ __launch_bounds__(128) void k_pad_nhwc(const float* __restrict__ x,
                                                  unsigned short* __restrict__ xp) {
  int job = blockIdx.x * 128 + threadIdx.x;
  int c   = job & 63;
  int pix = job >> 6;
  int b   = pix / (HP * WP);
  int rem = pix - b * (HP * WP);
  int y   = rem / WP;
  int xc  = rem - y * WP;
  float v = 0.f;
  if (y >= 1 && y <= HH && xc >= 1 && xc <= WW)
    v = x[(((size_t)(b * CIN + c)) * HH + (y - 1)) * WW + (xc - 1)];
  xp[(size_t)pix * 64 + c] = f2bf(v);
}

// --------------------------------------------------------------------------
// Kernel 2: swizzle weights into WMMA B-fragment order.
//  bf16 16x16x32 B-frag: lane L holds col N = L&15; elem e = K=(e>>3)*16+(L>>4)*8+(e&7).
//  Stored [ktile][ntile][lane][16];  K = tap*64 + c  (tap = ky*3+kx).
// --------------------------------------------------------------------------
__global__ __launch_bounds__(256) void k_prep_w(const float* __restrict__ pw,
                                                const float* __restrict__ cw,
                                                unsigned short* __restrict__ wOff,
                                                unsigned short* __restrict__ wMain) {
  int i = blockIdx.x * 256 + threadIdx.x;
  if (i < 18432) {                               // offset conv: 18 kt * 2 nt
    int kt = i >> 10;  int rem = i & 1023;
    int nt = rem >> 9; int r2 = rem & 511;
    int lane = r2 >> 4; int e = r2 & 15;
    int K = kt * 32 + ((e >> 3) << 4) + ((lane >> 4) << 3) + (e & 7);
    int n = nt * 16 + (lane & 15);
    int tap = K >> 6, c = K & 63;
    float v = 0.f;
    if (n < 18) v = pw[((n * 64 + c) * 3 + tap / 3) * 3 + (tap % 3)];
    wOff[i] = f2bf(v);
  } else if (i < 18432 + 36864) {                // main conv: 18 kt * 4 nt
    int j = i - 18432;
    int kt = j >> 11;  int rem = j & 2047;
    int nt = rem >> 9; int r2 = rem & 511;
    int lane = r2 >> 4; int e = r2 & 15;
    int K = kt * 32 + ((e >> 3) << 4) + ((lane >> 4) << 3) + (e & 7);
    int n = nt * 16 + (lane & 15);
    int tap = K >> 6, c = K & 63;
    wMain[j] = f2bf(cw[((n * 64 + c) * 3 + tap / 3) * 3 + (tap % 3)]);
  }
}

// --------------------------------------------------------------------------
// Kernel 3: offset conv (implicit GEMM). WG = 16 pixels.
// A-tile [16 px][576] bf16 built by async global->LDS 16B copies
// (pure data movement -> ASYNCcnt path). Waves 0,1 compute N-tiles.
// --------------------------------------------------------------------------
__global__ __launch_bounds__(128) void k_offconv(const unsigned short* __restrict__ xp,
                                                 const unsigned short* __restrict__ wOff,
                                                 const float* __restrict__ bias,
                                                 float* __restrict__ offbuf) {
  __shared__ unsigned short At[16 * KTOT];
  int wg  = blockIdx.x;
  int b   = wg >> 12;
  int rem = wg & 4095;
  int r   = rem >> 4;
  int s0  = (rem & 15) << 4;
  int tid = threadIdx.x;
  int g   = tid >> 3;          // 16 groups of 8 lanes; each copies 16B
  int l8  = tid & 7;

  // 144 (pixel,tap) jobs * 128B each = 16 groups x 9 iters x 16B
  for (int it = 0; it < 9; ++it) {
    int job = it * 16 + g;
    int p = job / 9, tap = job - p * 9;
    int ky = tap / 3, kx = tap - ky * 3;
    const unsigned short* src = xp + (((size_t)(b * HP + (r + ky))) * WP + (s0 + p + kx)) * 64 + l8 * 8;
    async_cp16(src, &At[p * KTOT + tap * 64 + l8 * 8]);
  }
  wait_async0();
  __syncthreads();

  int wave = tid >> 5, lane = tid & 31;
  if (wave < 2) {
    v8f acc = {};
    int row = lane & 15, half = lane >> 4;
    #pragma unroll
    for (int kt = 0; kt < KT; ++kt) {
      Frag A, B;
      A.q[0] = *(const uint4*)&At[row * KTOT + kt * 32 + half * 8];
      A.q[1] = *(const uint4*)&At[row * KTOT + kt * 32 + 16 + half * 8];
      const uint4* bp = (const uint4*)(wOff + (((kt * 2 + wave) * 32 + lane) << 4));
      B.q[0] = bp[0]; B.q[1] = bp[1];
      acc = __builtin_amdgcn_wmma_f32_16x16x32_bf16(false, A.v, false, B.v,
                                                    (short)0, acc, false, false);
    }
    int n = wave * 16 + (lane & 15);
    if (n < 18) {
      float bv = bias[n];
      int mh = (lane >> 4) * 8;
      #pragma unroll
      for (int i = 0; i < 8; ++i) {
        int p = i + mh;
        offbuf[(((size_t)(b * HH + r)) * WW + s0 + p) * 18 + n] = acc[i] + bv;
      }
    }
  }
}

// --------------------------------------------------------------------------
// Kernel 4: fused bilinear sampling + main GEMM.  WG = 32 pixels (M=32).
// Each wave owns one 16-wide N-tile and 2 M-subtiles: B-fragment loaded once
// per K-step, reused by 2 WMMAs. Outputs stored as float4 (16B contiguous).
// Position math replicates the reference exactly (incl. its row/col quirk):
//   row-in-xp = s+1 + a[n%3] + off[n],  col-in-xp = r+1 + a[n/3] + off[9+n]
// --------------------------------------------------------------------------
__global__ __launch_bounds__(128) void k_main(const unsigned short* __restrict__ xp,
                                              const unsigned short* __restrict__ wMain,
                                              const float* __restrict__ offbuf,
                                              float* __restrict__ out) {
  __shared__ unsigned short At[32 * KTOT];   // 36,864 B
  __shared__ float offs[32 * 18];            //  2,304 B
  int wg  = blockIdx.x;
  int b   = wg >> 11;
  int rem = wg & 2047;
  int r   = rem >> 3;
  int s0  = (rem & 7) << 5;
  int tid = threadIdx.x;

  // stage 32 pixels x 18 offsets (2304B, 16B-aligned) via async copies
  const float* ob = offbuf + (((size_t)(b * HH + r)) * WW + s0) * 18;
  #pragma unroll
  for (int it = 0; it < 2; ++it) {
    int idx = it * 128 + tid;
    if (idx < 144) async_cp16(ob + idx * 4, &offs[idx * 4]);
  }
  wait_async0();
  __syncthreads();

  int hw = tid >> 4, l16 = tid & 15;
  const float LIM = 257.f;
  size_t xbase = (size_t)b * (HP * WP * 64);
  for (int it = 0; it < 36; ++it) {          // 32 px * 9 taps / 8 half-waves
    int job = it * 8 + hw;
    int p = job / 9, tap = job - p * 9;
    float u = (float)(s0 + p + 1) + ((float)(tap % 3) - 1.f) + offs[p * 18 + tap];
    float v = (float)(r + 1)      + ((float)(tap / 3) - 1.f) + offs[p * 18 + 9 + tap];
    float fu = floorf(u), fv = floorf(v);
    float u0 = fminf(fmaxf(fu, 0.f), LIM),       v0 = fminf(fmaxf(fv, 0.f), LIM);
    float u1 = fminf(fmaxf(fu + 1.f, 0.f), LIM), v1 = fminf(fmaxf(fv + 1.f, 0.f), LIM);
    float uc = fminf(fmaxf(u, 0.f), LIM),        vc = fminf(fmaxf(v, 0.f), LIM);
    float g_lt = (1.f + (u0 - uc)) * (1.f + (v0 - vc));
    float g_rb = (1.f - (u1 - uc)) * (1.f - (v1 - vc));
    float g_lb = (1.f + (u0 - uc)) * (1.f - (v1 - vc));
    float g_rt = (1.f - (u1 - uc)) * (1.f + (v0 - vc));
    int iu0 = (int)u0, iu1 = (int)u1, iv0 = (int)v0, iv1 = (int)v1;
    uint2 qlt = *(const uint2*)(xp + xbase + ((size_t)iu0 * WP + iv0) * 64 + l16 * 4);
    uint2 qrb = *(const uint2*)(xp + xbase + ((size_t)iu1 * WP + iv1) * 64 + l16 * 4);
    uint2 qlb = *(const uint2*)(xp + xbase + ((size_t)iu0 * WP + iv1) * 64 + l16 * 4);
    uint2 qrt = *(const uint2*)(xp + xbase + ((size_t)iu1 * WP + iv0) * 64 + l16 * 4);
    float r0 = g_lt * bf2f(qlt.x)       + g_rb * bf2f(qrb.x)       + g_lb * bf2f(qlb.x)       + g_rt * bf2f(qrt.x);
    float r1 = g_lt * bf2f(qlt.x >> 16) + g_rb * bf2f(qrb.x >> 16) + g_lb * bf2f(qlb.x >> 16) + g_rt * bf2f(qrt.x >> 16);
    float r2 = g_lt * bf2f(qlt.y)       + g_rb * bf2f(qrb.y)       + g_lb * bf2f(qlb.y)       + g_rt * bf2f(qrt.y);
    float r3 = g_lt * bf2f(qlt.y >> 16) + g_rb * bf2f(qrb.y >> 16) + g_lb * bf2f(qlb.y >> 16) + g_rt * bf2f(qrt.y >> 16);
    *(uint2*)&At[p * KTOT + tap * 64 + l16 * 4] = make_uint2(pack2(r0, r1), pack2(r2, r3));
  }
  __syncthreads();

  int wave = tid >> 5, lane = tid & 31;
  int row = lane & 15, half = lane >> 4;
  __builtin_prefetch(wMain, 0, 3);           // global_prefetch_b8: hot weights
  v8f acc0 = {}, acc1 = {};
  #pragma unroll
  for (int kt = 0; kt < KT; ++kt) {
    Frag A0, A1, B;
    const uint4* bp = (const uint4*)(wMain + (((kt * 4 + wave) * 32 + lane) << 4));
    B.q[0] = bp[0]; B.q[1] = bp[1];
    A0.q[0] = *(const uint4*)&At[row * KTOT + kt * 32 + half * 8];
    A0.q[1] = *(const uint4*)&At[row * KTOT + kt * 32 + 16 + half * 8];
    A1.q[0] = *(const uint4*)&At[(16 + row) * KTOT + kt * 32 + half * 8];
    A1.q[1] = *(const uint4*)&At[(16 + row) * KTOT + kt * 32 + 16 + half * 8];
    acc0 = __builtin_amdgcn_wmma_f32_16x16x32_bf16(false, A0.v, false, B.v,
                                                   (short)0, acc0, false, false);
    acc1 = __builtin_amdgcn_wmma_f32_16x16x32_bf16(false, A1.v, false, B.v,
                                                   (short)0, acc1, false, false);
  }
  int n  = wave * 16 + (lane & 15);
  int mh = half * 8;
  size_t obase2 = (((size_t)(b * 64 + n)) * HH + r) * WW + s0;
  float4 lo0 = make_float4(acc0[0], acc0[1], acc0[2], acc0[3]);
  float4 hi0 = make_float4(acc0[4], acc0[5], acc0[6], acc0[7]);
  float4 lo1 = make_float4(acc1[0], acc1[1], acc1[2], acc1[3]);
  float4 hi1 = make_float4(acc1[4], acc1[5], acc1[6], acc1[7]);
  *(float4*)&out[obase2 + mh + 0]      = lo0;   // M-subtile 0: pixels mh..mh+7
  *(float4*)&out[obase2 + mh + 4]      = hi0;
  *(float4*)&out[obase2 + 16 + mh + 0] = lo1;   // M-subtile 1
  *(float4*)&out[obase2 + 16 + mh + 4] = hi1;
}

// --------------------------------------------------------------------------
extern "C" void kernel_launch(void* const* d_in, const int* in_sizes, int n_in,
                              void* d_out, int out_size, void* d_ws, size_t ws_size,
                              hipStream_t stream) {
  const float* x  = (const float*)d_in[0];   // (2,64,256,256) f32
  const float* pw = (const float*)d_in[1];   // (18,64,3,3)    f32
  const float* pb = (const float*)d_in[2];   // (18,)          f32
  const float* cw = (const float*)d_in[3];   // (64,64,3,3)    f32
  float* out = (float*)d_out;                // (2,64,256,256) f32

  char* ws = (char*)d_ws;
  unsigned short* xp    = (unsigned short*)ws;                      // 17,040,384 B
  unsigned short* wOff  = (unsigned short*)(ws + 17040384);         //     36,864 B
  unsigned short* wMain = (unsigned short*)(ws + 17040384 + 36864); //     73,728 B
  float* offbuf = (float*)(ws + 17040384 + 36864 + 73728);          //  9,437,184 B

  k_pad_nhwc<<<66564, 128, 0, stream>>>(x, xp);
  k_prep_w  <<<216,   256, 0, stream>>>(pw, cw, wOff, wMain);
  k_offconv <<<8192,  128, 0, stream>>>(xp, wOff, pb, offbuf);
  k_main    <<<4096,  128, 0, stream>>>(xp, wMain, offbuf, out);
}
